// QuantumCritic_14585708937275
// MI455X (gfx1250) — matrix operand order repaired and checked by reading, not compiled, not run on hardware
//
#include <hip/hip_runtime.h>
#include <hip/hip_bf16.h>

typedef _Float16 v16h __attribute__((ext_vector_type(16)));
typedef _Float16 h8   __attribute__((ext_vector_type(8)));
typedef float    v8f  __attribute__((ext_vector_type(8)));

union V16 { v16h v; h8 h[2]; };

#define NQ 4
#define NL 2

// ---------------------------------------------------------------------------
// Prep kernel 1: convert W1 (f32 [256,160]) of both branches into f16
// fragments laid out [b][kt(5)][nt(16)][lane(32)][e(16)] so the main kernel
// loads each B-fragment as two coalesced b128 per lane.
// B-matrix 32x16 f16 layout: lane%16 = N column; lanes<16 hold K=kt*32+0..7,16..23,
// lanes>=16 hold K=kt*32+8..15,24..31 (elements 0..7 then 8..15).
// ---------------------------------------------------------------------------
__global__ void prep_w1frags(const float* __restrict__ W1a,
                             const float* __restrict__ W1b,
                             _Float16* __restrict__ frag) {
  const int total = 2 * 5 * 16 * 32 * 16;  // 81920
  for (int f = blockIdx.x * blockDim.x + threadIdx.x; f < total;
       f += gridDim.x * blockDim.x) {
    int b    = f / 40960;
    int r    = f - b * 40960;
    int kt   = r >> 13;          // /8192
    int nt   = (r >> 9) & 15;
    int lane = (r >> 4) & 31;
    int e    = r & 15;
    const float* W = b ? W1b : W1a;
    int n = nt * 16 + (lane & 15);
    int k = kt * 32 + ((lane < 16) ? 0 : 8) + (e & 7) + ((e & 8) ? 16 : 0);
    frag[f] = (_Float16)W[n * 160 + k];
  }
}

// ---------------------------------------------------------------------------
// Prep kernel 2: build A = Re(U^H M U) (16x16 real symmetric) per branch,
// where U = fixed part of the circuit (RX/RY/RZ layers + CNOT ring) and
// M = diag(+1 for idx<8, -1 for idx>=8)  (qubit 0 is the MSB).
// 32 threads: thread (b,j) simulates circuit on basis column j.
// ---------------------------------------------------------------------------
__global__ void build_A(const float* __restrict__ qw1,
                        const float* __restrict__ qw2,
                        float* __restrict__ Aout) {
  __shared__ float Ur[2][16][16];  // [b][k][j]
  __shared__ float Ui[2][16][16];
  int tid = threadIdx.x;
  int b = tid >> 4, j = tid & 15;
  const float* qw = b ? qw2 : qw1;  // [2][4][3]

  float re[16], im[16];
  for (int k = 0; k < 16; ++k) { re[k] = (k == j) ? 1.f : 0.f; im[k] = 0.f; }

  for (int l = 0; l < NL; ++l) {
    for (int q = 0; q < NQ; ++q) {
      int m = 8 >> q;
      float th, c, s;
      // RX: [[c, -is], [-is, c]]
      th = qw[(l * NQ + q) * 3 + 0]; c = cosf(0.5f * th); s = sinf(0.5f * th);
      for (int idx = 0; idx < 16; ++idx) if (!(idx & m)) {
        int i1 = idx | m;
        float r0 = re[idx], i0 = im[idx], r1 = re[i1], i1m = im[i1];
        re[idx] = c * r0 + s * i1m;  im[idx] = c * i0 - s * r1;
        re[i1]  = s * i0 + c * r1;   im[i1]  = -s * r0 + c * i1m;
      }
      // RY: [[c, -s], [s, c]] (real)
      th = qw[(l * NQ + q) * 3 + 1]; c = cosf(0.5f * th); s = sinf(0.5f * th);
      for (int idx = 0; idx < 16; ++idx) if (!(idx & m)) {
        int i1 = idx | m;
        float r0 = re[idx], i0 = im[idx], r1 = re[i1], i1m = im[i1];
        re[idx] = c * r0 - s * r1;  im[idx] = c * i0 - s * i1m;
        re[i1]  = s * r0 + c * r1;  im[i1]  = s * i0 + c * i1m;
      }
      // RZ: diag(e^{-i th/2}, e^{+i th/2})
      th = qw[(l * NQ + q) * 3 + 2]; c = cosf(0.5f * th); s = sinf(0.5f * th);
      for (int idx = 0; idx < 16; ++idx) if (!(idx & m)) {
        int i1 = idx | m;
        float r0 = re[idx], i0 = im[idx], r1 = re[i1], i1m = im[i1];
        re[idx] = c * r0 + s * i0;   im[idx] = c * i0 - s * r0;
        re[i1]  = c * r1 - s * i1m;  im[i1]  = c * i1m + s * r1;
      }
    }
    // CNOT(0,1), CNOT(1,2), CNOT(2,3), CNOT(3,0)
    for (int e = 0; e < NQ; ++e) {
      int cq = e, tq = (e + 1) & 3;
      int mc = 8 >> cq, mt = 8 >> tq;
      for (int idx = 0; idx < 16; ++idx)
        if ((idx & mc) && !(idx & mt)) {
          int i1 = idx | mt;
          float t0 = re[idx]; re[idx] = re[i1]; re[i1] = t0;
          float t1 = im[idx]; im[idx] = im[i1]; im[i1] = t1;
        }
    }
  }
  for (int k = 0; k < 16; ++k) { Ur[b][k][j] = re[k]; Ui[b][k][j] = im[k]; }
  __syncthreads();
  for (int i = 0; i < 16; ++i) {
    float a = 0.f;
    for (int k = 0; k < 16; ++k) {
      float t = Ur[b][k][i] * Ur[b][k][j] + Ui[b][k][i] * Ui[b][k][j];
      a += (k < 8) ? t : -t;
    }
    Aout[b * 256 + i * 16 + j] = a;
  }
}

// ---------------------------------------------------------------------------
// Main fused kernel: 256 threads (8 waves), 128 rows per block.
// Each wave owns a 16-row tile: WMMA GEMM1 (K=160 via 5 k-frags resident in
// VGPRs, 16 n-tiles), fused bias+ReLU+GEMM2 epilogue in VALU, shfl butterfly
// reduction, then per-row 4-qubit expval v^T A v and the final affine.
// ---------------------------------------------------------------------------
__global__ __launch_bounds__(256) void critic_kernel(
    const float* __restrict__ state, const float* __restrict__ action,
    const _Float16* __restrict__ w1frag, const float* __restrict__ Aws,
    const float* __restrict__ b1_0, const float* __restrict__ W2_0,
    const float* __restrict__ b2_0, const float* __restrict__ pw_0,
    const float* __restrict__ pb_0,
    const float* __restrict__ b1_1, const float* __restrict__ W2_1,
    const float* __restrict__ b2_1, const float* __restrict__ pw_1,
    const float* __restrict__ pb_1,
    float* __restrict__ dout, int Btot) {
  __shared__ __align__(16) _Float16 xlds[128 * 168];
  __shared__ float W2lds[4 * 256];
  __shared__ float b1lds[256];
  __shared__ float Alds[256];
  __shared__ float qlds[8][16][4];

  const int tid = threadIdx.x;
  const int lane = tid & 31;
  const int waveId = tid >> 5;
  const int rowBase = blockIdx.x * 128;

  // ---- stage x = concat(state,action) rows -> LDS as f16 ----
  for (int i = tid; i < 128 * 160; i += 256) {
    int r = i / 160, c = i - r * 160;
    int grow = rowBase + r;
    if (grow >= Btot) grow = Btot - 1;
    float val = (c < 128) ? state[(size_t)grow * 128 + c]
                          : action[(size_t)grow * 32 + (c - 128)];
    xlds[r * 168 + c] = (_Float16)val;
  }

  for (int b = 0; b < 2; ++b) {
    __syncthreads();  // x staged (b=0) / previous branch fully consumed (b=1)
    const float* b1p = b ? b1_1 : b1_0;
    const float* W2p = b ? W2_1 : W2_0;
    const float* b2p = b ? b2_1 : b2_0;
    const float pwv = b ? pw_1[0] : pw_0[0];
    const float pbv = b ? pb_1[0] : pb_0[0];
    for (int i = tid; i < 1024; i += 256) W2lds[i] = W2p[i];
    b1lds[tid] = b1p[tid];
    Alds[tid] = Aws[b * 256 + tid];
    __syncthreads();

    // ---- A fragments: 16x32 f16, 5 k-tiles, resident in VGPRs ----
    const int lrow = waveId * 16 + (lane & 15);
    const int koff = (lane < 16) ? 0 : 8;
    V16 af[5];
#pragma unroll
    for (int kt = 0; kt < 5; ++kt) {
      int base = lrow * 168 + kt * 32 + koff;
      af[kt].h[0] = *(const h8*)&xlds[base];
      af[kt].h[1] = *(const h8*)&xlds[base + 16];
    }

    float pq[8][4];
#pragma unroll
    for (int v = 0; v < 8; ++v)
#pragma unroll
      for (int q = 0; q < 4; ++q) pq[v][q] = 0.f;

    const int col0 = lane & 15;
    for (int nt = 0; nt < 16; ++nt) {
      v8f acc = {};
#pragma unroll
      for (int kt = 0; kt < 5; ++kt) {
        const h8* wp = (const h8*)(w1frag +
            ((((size_t)(b * 5 + kt) * 16 + nt) * 32 + lane) << 4));
        V16 bf; bf.h[0] = wp[0]; bf.h[1] = wp[1];
        acc = __builtin_amdgcn_wmma_f32_16x16x32_f16(
            false, af[kt].v, false, bf.v, (short)0, acc, false, false);
      }
      const int col = nt * 16 + col0;
      const float bias = b1lds[col];
      const float w2c0 = W2lds[0 * 256 + col];
      const float w2c1 = W2lds[1 * 256 + col];
      const float w2c2 = W2lds[2 * 256 + col];
      const float w2c3 = W2lds[3 * 256 + col];
#pragma unroll
      for (int v = 0; v < 8; ++v) {
        float hv = fmaxf(acc[v] + bias, 0.f);
        pq[v][0] += hv * w2c0;
        pq[v][1] += hv * w2c1;
        pq[v][2] += hv * w2c2;
        pq[v][3] += hv * w2c3;
      }
    }

    // ---- butterfly reduce over the 16 columns within each half-wave ----
#pragma unroll
    for (int off = 1; off < 16; off <<= 1)
#pragma unroll
      for (int v = 0; v < 8; ++v)
#pragma unroll
        for (int q = 0; q < 4; ++q)
          pq[v][q] += __shfl_xor(pq[v][q], off, 32);

    // lanes 0 and 16 publish qin (+b2) for rows 0..7 / 8..15 of the tile
    if ((lane & 15) == 0) {
      int rb = (lane >> 4) * 8;
#pragma unroll
      for (int v = 0; v < 8; ++v)
#pragma unroll
        for (int q = 0; q < 4; ++q)
          qlds[waveId][rb + v][q] = pq[v][q] + b2p[q];
    }
    __syncthreads();

    // ---- per-row quantum expval: <Z0> = v^T A v ----
    if (lane < 16) {
      float t0 = qlds[waveId][lane][0];
      float t1 = qlds[waveId][lane][1];
      float t2 = qlds[waveId][lane][2];
      float t3 = qlds[waveId][lane][3];
      float c0 = __cosf(0.5f * t0), s0 = __sinf(0.5f * t0);
      float c1 = __cosf(0.5f * t1), s1 = __sinf(0.5f * t1);
      float c2 = __cosf(0.5f * t2), s2 = __sinf(0.5f * t2);
      float c3 = __cosf(0.5f * t3), s3 = __sinf(0.5f * t3);
      float vv[16];
#pragma unroll
      for (int idx = 0; idx < 16; ++idx)
        vv[idx] = ((idx & 8) ? s0 : c0) * ((idx & 4) ? s1 : c1) *
                  ((idx & 2) ? s2 : c2) * ((idx & 1) ? s3 : c3);
      float ev = 0.f;
#pragma unroll
      for (int i = 0; i < 16; ++i) {
        float u = 0.f;
#pragma unroll
        for (int jj = 0; jj < 16; ++jj) u += Alds[i * 16 + jj] * vv[jj];
        ev += vv[i] * u;
      }
      int grow = rowBase + waveId * 16 + lane;
      if (grow < Btot) dout[(size_t)b * Btot + grow] = ev * pwv + pbv;
    }
  }
}

extern "C" void kernel_launch(void* const* d_in, const int* in_sizes, int n_in,
                              void* d_out, int out_size, void* d_ws,
                              size_t ws_size, hipStream_t stream) {
  const float* state  = (const float*)d_in[0];
  const float* action = (const float*)d_in[1];
  const float* q1_W1 = (const float*)d_in[2];
  const float* q1_b1 = (const float*)d_in[3];
  const float* q1_W2 = (const float*)d_in[4];
  const float* q1_b2 = (const float*)d_in[5];
  const float* q2_W1 = (const float*)d_in[6];
  const float* q2_b1 = (const float*)d_in[7];
  const float* q2_W2 = (const float*)d_in[8];
  const float* q2_b2 = (const float*)d_in[9];
  const float* q1_qw = (const float*)d_in[10];
  const float* q2_qw = (const float*)d_in[11];
  const float* q1_pw = (const float*)d_in[12];
  const float* q1_pb = (const float*)d_in[13];
  const float* q2_pw = (const float*)d_in[14];
  const float* q2_pb = (const float*)d_in[15];

  int Btot = in_sizes[0] / 128;  // state is [B,128]

  float* Aws = (float*)d_ws;                               // 2*256 f32
  _Float16* frag = (_Float16*)((char*)d_ws + 2048);        // 2*5*16*32*16 f16

  prep_w1frags<<<320, 256, 0, stream>>>(q1_W1, q2_W1, frag);
  build_A<<<1, 32, 0, stream>>>(q1_qw, q2_qw, Aws);

  int grid = (Btot + 127) / 128;
  critic_kernel<<<grid, 256, 0, stream>>>(
      state, action, frag, Aws,
      q1_b1, q1_W2, q1_b2, q1_pw, q1_pb,
      q2_b1, q2_W2, q2_b2, q2_pw, q2_pb,
      (float*)d_out, Btot);
}
